// Seq2Seq_1331439862420
// MI455X (gfx1250) — matrix operand order repaired and checked by reading, not compiled, not run on hardware
//
#include <hip/hip_runtime.h>

// ---------------------------------------------------------------------------
// Seq2Seq LSTM (B=64, S=T=64, E=256, U=512, V=32000) for MI455X (gfx1250).
//   1. Weights -> f16; embeddings gathered into f16 (t,b,e).
//   2. Time-parallel Xg[t] = xe_t @ Wih^T + bih + bhh (M=4096,N=2048,K=256).
//   3. Sequential recurrence: gates = Xg[t] + h @ Whh^T (M=64,N=2048,K=512)
//      + fused sigmoid/tanh elementwise (h re-emitted as f16 WMMA A-operand).
//   4. Logits GEMM M=4096,N=32000,K=512 (134 GFLOP, dominant), out_W f16
//      (32.8 MB, L2-resident in 192 MB L2).
// All GEMMs: v_wmma_f32_16x16x32_f16; B-operand strip staged into LDS once
// per block (shared by all waves) via GLOBAL_LOAD_ASYNC_TO_LDS_B128 when the
// toolchain exposes it, else a plain b128 global->LDS copy.
// ---------------------------------------------------------------------------

typedef __attribute__((ext_vector_type(16))) _Float16 v16h;
typedef __attribute__((ext_vector_type(8)))  float    v8f;
typedef int __attribute__((ext_vector_type(4))) i4v;

#define B_ 64
#define S_ 64
#define T_ 64
#define E_ 256
#define U_ 512
#define G_ (4 * U_)   // 2048 gate width
#define V_ 32000

#if defined(__has_builtin)
#if __has_builtin(__builtin_amdgcn_global_load_async_to_lds_b128)
#define HAVE_ASYNC_LDS 1
#endif
#endif
#ifndef HAVE_ASYNC_LDS
#define HAVE_ASYNC_LDS 0
#endif

#if HAVE_ASYNC_LDS
typedef __attribute__((address_space(1))) i4v* gbl_i4p;   // global int4*
typedef __attribute__((address_space(3))) i4v* lds_i4p;   // LDS int4*
#endif

// ---------------------------------------------------------------------------
// Small utility kernels
// ---------------------------------------------------------------------------

__global__ void f32_to_f16_kernel(const float* __restrict__ src,
                                  _Float16* __restrict__ dst, int n) {
  int i = blockIdx.x * blockDim.x + threadIdx.x;
  if (i < n) dst[i] = (_Float16)src[i];
}

// Encoder embeddings: out[t][b][e] = enc_emb[x[b*S+t]][e]   (f16)
__global__ void embed_enc_kernel(const int* __restrict__ x,
                                 const float* __restrict__ emb,
                                 _Float16* __restrict__ out) {
  int idx = blockIdx.x * blockDim.x + threadIdx.x;   // S*B*E threads
  int e = idx & (E_ - 1);
  int rem = idx >> 8;                                // /E_
  int b = rem & (B_ - 1);
  int t = rem >> 6;                                  // /B_
  int tok = x[b * S_ + t];
  out[idx] = (_Float16)emb[(size_t)tok * E_ + e];
}

// Decoder embeddings: out[t][b][e] = dec_emb[y[b*(T+1)+t]][e]   (f16)
__global__ void embed_dec_kernel(const int* __restrict__ y,
                                 const float* __restrict__ emb,
                                 _Float16* __restrict__ out) {
  int idx = blockIdx.x * blockDim.x + threadIdx.x;   // T*B*E threads
  int e = idx & (E_ - 1);
  int rem = idx >> 8;
  int b = rem & (B_ - 1);
  int t = rem >> 6;
  int tok = y[b * (T_ + 1) + t];
  out[idx] = (_Float16)emb[(size_t)tok * E_ + e];
}

__global__ void init_state_kernel(float* __restrict__ c,
                                  _Float16* __restrict__ h_h) {
  int idx = blockIdx.x * blockDim.x + threadIdx.x;   // B*U threads
  c[idx] = 0.0f;
  h_h[idx] = (_Float16)0.0f;
}

// ---------------------------------------------------------------------------
// WMMA fragment helpers (ISA 7.12.2 layouts, wave32)
// ---------------------------------------------------------------------------

// A tile: 16x32 f16, rows m0..m0+15 of row-major A (lda elems), K chunk k0..k0+31.
__device__ __forceinline__ v16h load_a_frag(const _Float16* __restrict__ A,
                                            int lda, int m0, int k0, int lane) {
  int r = lane & 15, hf = lane >> 4;
  const _Float16* row = A + (size_t)(m0 + r) * lda;
  v16h a;
#pragma unroll
  for (int i = 0; i < 4; ++i) {
    int k = k0 + hf * 16 + 2 * i;
    a[2 * i]     = row[k];
    a[2 * i + 1] = row[k + 1];
    int k2 = k0 + hf * 16 + 8 + 2 * i;
    a[8 + 2 * i]     = row[k2];
    a[8 + 2 * i + 1] = row[k2 + 1];
  }
  return a;
}

// LDS-staged B strip: 64 cols x 32 K halves, padded row stride (bank spread).
constexpr int BSTRIDE = 40;   // halves; 80B row stride -> 16 distinct banks

// B frag from LDS strip: col = n_local + (lane&15); lane half selects K 0..15
// vs 16..31 -> one contiguous 32B read per lane.
__device__ __forceinline__ v16h load_b_frag_lds(const _Float16* S, int n_local,
                                                int lane) {
  int c = lane & 15, hf = lane >> 4;
  const _Float16* p = S + (n_local + c) * BSTRIDE + hf * 16;
  return *(const v16h*)p;
}

__device__ __forceinline__ v8f wmma_f16(v16h a, v16h b, v8f c) {
  return __builtin_amdgcn_wmma_f32_16x16x32_f16(
      /*neg_a=*/false, a, /*neg_b=*/false, b,
      /*c_mod=*/(short)0, c, /*reuse_a=*/false, /*reuse_b=*/false);
}

// 16-byte global -> LDS stage (async when available).
__device__ __forceinline__ void stage16(const _Float16* __restrict__ g,
                                        _Float16* l) {
#if HAVE_ASYNC_LDS
  __builtin_amdgcn_global_load_async_to_lds_b128((gbl_i4p)g, (lds_i4p)l, 0, 0);
#else
  *(i4v*)l = *(const i4v*)g;
#endif
}

__device__ __forceinline__ void wait_async0() {
#if HAVE_ASYNC_LDS
#if defined(__has_builtin) && __has_builtin(__builtin_amdgcn_s_wait_asynccnt)
  __builtin_amdgcn_s_wait_asynccnt(0);
#else
  asm volatile("s_wait_asynccnt 0" ::: "memory");
#endif
#endif
}

// ---------------------------------------------------------------------------
// Unified staged WMMA GEMM: out[m,n] = sum_k A[m,k] * W[n,k] + epilogue.
// Block = WM waves. All waves share one 64-col B strip (staged in LDS once);
// wave w computes m-tile (m_group*WM + w), 4 n-tiles of 16.
// EPI: 0 = +p0[n]+p1[n] (bias pair)     -> Xg precompute
//      1 = +p0[m,n]     (add matrix)    -> recurrent gates
//      2 = +p0[n], rows m=t*B+b scatter-stored to out[b][t][n]  -> logits
// ---------------------------------------------------------------------------
template <int WM, int EPI>
__global__ __launch_bounds__(32 * WM) void wmma_gemm_staged(
    const _Float16* __restrict__ A, const _Float16* __restrict__ W,
    const float* __restrict__ p0, const float* __restrict__ p1,
    float* __restrict__ out, int M, int N, int K) {
  __shared__ _Float16 bs[64 * BSTRIDE];

  const int tid = threadIdx.x;
  const int wave = tid >> 5, lane = tid & 31;
  const int n_groups = N >> 6;
  const int n_base = (blockIdx.x % n_groups) << 6;
  const int m0 = ((blockIdx.x / n_groups) * WM + wave) << 4;

  const _Float16* arow = A + (size_t)(m0 + (lane & 15)) * K;

  v8f acc0 = {}, acc1 = {}, acc2 = {}, acc3 = {};
  for (int k0 = 0; k0 < K; k0 += 32) {
    // Cooperative stage of W[n_base..n_base+64) x [k0..k0+32) into LDS:
    // 256 16-byte chunks; chunk i -> (col = i/4, K-octet = i%4).
    for (int i = tid; i < 256; i += 32 * WM) {
      int col = i >> 2, oct = i & 3;
      stage16(W + (size_t)(n_base + col) * K + (k0 + oct * 8),
              &bs[col * BSTRIDE + oct * 8]);
    }
    wait_async0();
    __syncthreads();

    __builtin_prefetch(arow + k0 + 64, 0, 0);   // next A chunk -> L2/L0
    v16h a = load_a_frag(A, K, m0, k0, lane);
    acc0 = wmma_f16(a, load_b_frag_lds(bs,  0, lane), acc0);
    acc1 = wmma_f16(a, load_b_frag_lds(bs, 16, lane), acc1);
    acc2 = wmma_f16(a, load_b_frag_lds(bs, 32, lane), acc2);
    acc3 = wmma_f16(a, load_b_frag_lds(bs, 48, lane), acc3);
    __syncthreads();
  }

  int r = lane & 15, hf = lane >> 4;
  v8f accs[4] = {acc0, acc1, acc2, acc3};
#pragma unroll
  for (int tile = 0; tile < 4; ++tile) {
    int col = n_base + tile * 16 + r;
    if constexpr (EPI == 0) {
      float bsum = p0[col] + p1[col];
#pragma unroll
      for (int j = 0; j < 8; ++j) {
        int row = m0 + hf * 8 + j;
        out[(size_t)row * N + col] = accs[tile][j] + bsum;
      }
    } else if constexpr (EPI == 1) {
#pragma unroll
      for (int j = 0; j < 8; ++j) {
        int row = m0 + hf * 8 + j;
        size_t o = (size_t)row * N + col;
        out[o] = accs[tile][j] + p0[o];
      }
    } else {  // EPI == 2: logits scatter
      float bv = p0[col];
#pragma unroll
      for (int j = 0; j < 8; ++j) {
        int m = m0 + hf * 8 + j;          // m = t*B + b
        int b = m & (B_ - 1);
        int t = m >> 6;
        out[((size_t)b * T_ + t) * V_ + col] = accs[tile][j] + bv;
      }
    }
  }
}

// ---------------------------------------------------------------------------
// Fused LSTM elementwise: gates (B x 4U, order i,f,g,o) -> c (f32, in/out),
// h (f16 for next-step WMMA A-operand), optional copy into hs history.
// ---------------------------------------------------------------------------
__device__ __forceinline__ float sigmoidf_(float x) {
  return 1.0f / (1.0f + __expf(-x));
}

__global__ void lstm_elem_kernel(const float* __restrict__ gates,
                                 float* __restrict__ c,
                                 _Float16* __restrict__ h_h,
                                 _Float16* __restrict__ hs_slot) {
  int idx = blockIdx.x * blockDim.x + threadIdx.x;   // B*U threads exactly
  int b = idx >> 9;                                  // /U_
  int u = idx & (U_ - 1);
  const float* g = gates + (size_t)b * G_;
  float ig = sigmoidf_(g[u]);
  float fg = sigmoidf_(g[U_ + u]);
  float gg = tanhf(g[2 * U_ + u]);
  float og = sigmoidf_(g[3 * U_ + u]);
  float cn = fg * c[idx] + ig * gg;
  float hn = og * tanhf(cn);
  c[idx] = cn;
  h_h[idx] = (_Float16)hn;
  if (hs_slot) hs_slot[idx] = (_Float16)hn;
}

// ---------------------------------------------------------------------------
// Host-side orchestration
// ---------------------------------------------------------------------------
extern "C" void kernel_launch(void* const* d_in, const int* in_sizes, int n_in,
                              void* d_out, int out_size, void* d_ws, size_t ws_size,
                              hipStream_t stream) {
  const int*   x       = (const int*)  d_in[0];
  const int*   y       = (const int*)  d_in[1];
  const float* enc_emb = (const float*)d_in[2];
  const float* enc_Wih = (const float*)d_in[3];
  const float* enc_Whh = (const float*)d_in[4];
  const float* enc_bih = (const float*)d_in[5];
  const float* enc_bhh = (const float*)d_in[6];
  const float* dec_emb = (const float*)d_in[7];
  const float* dec_Wih = (const float*)d_in[8];
  const float* dec_Whh = (const float*)d_in[9];
  const float* dec_bih = (const float*)d_in[10];
  const float* dec_bhh = (const float*)d_in[11];
  const float* out_W   = (const float*)d_in[12];
  const float* out_b   = (const float*)d_in[13];
  float* out = (float*)d_out;

  char* ws = (char*)d_ws;
  size_t off = 0;
  auto take = [&](size_t bytes) -> char* {
    char* p = ws + off;
    off += (bytes + 255) & ~(size_t)255;
    return p;
  };

  _Float16* encWih_h = (_Float16*)take((size_t)G_ * E_ * 2);
  _Float16* encWhh_h = (_Float16*)take((size_t)G_ * U_ * 2);
  _Float16* decWih_h = (_Float16*)take((size_t)G_ * E_ * 2);
  _Float16* decWhh_h = (_Float16*)take((size_t)G_ * U_ * 2);
  _Float16* outW_h   = (_Float16*)take((size_t)V_ * U_ * 2);
  _Float16* xe_h     = (_Float16*)take((size_t)S_ * B_ * E_ * 2);
  _Float16* de_h     = (_Float16*)take((size_t)T_ * B_ * E_ * 2);
  float*    Xg_enc   = (float*)   take((size_t)S_ * B_ * G_ * 4);
  float*    Xg_dec   = (float*)   take((size_t)T_ * B_ * G_ * 4);
  float*    gates    = (float*)   take((size_t)B_ * G_ * 4);
  float*    c_buf    = (float*)   take((size_t)B_ * U_ * 4);
  _Float16* h_h      = (_Float16*)take((size_t)B_ * U_ * 2);
  _Float16* hs_h     = (_Float16*)take((size_t)T_ * B_ * U_ * 2);

  // 1. Weight conversions (f32 -> f16).
  f32_to_f16_kernel<<<(G_ * E_) / 256, 256, 0, stream>>>(enc_Wih, encWih_h, G_ * E_);
  f32_to_f16_kernel<<<(G_ * U_) / 256, 256, 0, stream>>>(enc_Whh, encWhh_h, G_ * U_);
  f32_to_f16_kernel<<<(G_ * E_) / 256, 256, 0, stream>>>(dec_Wih, decWih_h, G_ * E_);
  f32_to_f16_kernel<<<(G_ * U_) / 256, 256, 0, stream>>>(dec_Whh, decWhh_h, G_ * U_);
  f32_to_f16_kernel<<<(V_ * U_) / 256, 256, 0, stream>>>(out_W, outW_h, V_ * U_);

  // 2. Embedding gathers.
  embed_enc_kernel<<<(S_ * B_ * E_) / 256, 256, 0, stream>>>(x, enc_emb, xe_h);
  embed_dec_kernel<<<(T_ * B_ * E_) / 256, 256, 0, stream>>>(y, dec_emb, de_h);

  // 3. Time-parallel input GEMMs: Xg = xe @ Wih^T + bih + bhh.
  //    M=4096 N=2048 K=256; 8-wave blocks -> (4096/128)*(2048/64) = 1024 blocks.
  {
    int blocks = ((S_ * B_) / (16 * 8)) * (G_ / 64);
    wmma_gemm_staged<8, 0><<<blocks, 256, 0, stream>>>(
        xe_h, encWih_h, enc_bih, enc_bhh, Xg_enc, S_ * B_, G_, E_);
    wmma_gemm_staged<8, 0><<<blocks, 256, 0, stream>>>(
        de_h, decWih_h, dec_bih, dec_bhh, Xg_dec, T_ * B_, G_, E_);
  }

  // 4. Recurrence. h0 = c0 = 0; decoder continues from encoder final state.
  init_state_kernel<<<(B_ * U_) / 256, 256, 0, stream>>>(c_buf, h_h);

  //    M=64 N=2048 K=512; 4-wave blocks -> (64/64)*(2048/64) = 32 blocks.
  const int step_blocks = ((B_) / (16 * 4)) * (G_ / 64);
  for (int t = 0; t < S_; ++t) {
    wmma_gemm_staged<4, 1><<<step_blocks, 128, 0, stream>>>(
        h_h, encWhh_h, Xg_enc + (size_t)t * B_ * G_, nullptr, gates, B_, G_, U_);
    lstm_elem_kernel<<<(B_ * U_) / 256, 256, 0, stream>>>(
        gates, c_buf, h_h, (_Float16*)nullptr);
  }
  for (int t = 0; t < T_; ++t) {
    wmma_gemm_staged<4, 1><<<step_blocks, 128, 0, stream>>>(
        h_h, decWhh_h, Xg_dec + (size_t)t * B_ * G_, nullptr, gates, B_, G_, U_);
    lstm_elem_kernel<<<(B_ * U_) / 256, 256, 0, stream>>>(
        gates, c_buf, h_h, hs_h + (size_t)t * B_ * U_);
  }

  // 5. Logits: (T*B, U) @ out_W^T + out_b, scattered to (B, T, V).
  //    M=4096 N=32000 K=512; (4096/128)*(32000/64) = 16000 blocks.
  {
    int blocks = ((T_ * B_) / (16 * 8)) * (V_ / 64);
    wmma_gemm_staged<8, 2><<<blocks, 256, 0, stream>>>(
        hs_h, outW_h, out_b, nullptr, out, T_ * B_, V_, U_);
  }
}
